// HybridMoE_86277303042216
// MI455X (gfx1250) — matrix operand order repaired and checked by reading, not compiled, not run on hardware
//
#include <hip/hip_runtime.h>
#include <hip/hip_bf16.h>

#define NUM_EXPERTS 8
#define HIDDEN 2048
#define INTER 1408
#define TOKENS 4096

#define BM 128     // block M tile (tokens)
#define BN 64      // block N tile
#define KC 32      // K chunk per iteration
#define LDA 34     // padded LDS row stride in halfwords (matches TDM pad cfg)

typedef __attribute__((ext_vector_type(16))) __bf16 v16bf;
typedef __attribute__((ext_vector_type(8)))  float  v8f;
typedef __attribute__((ext_vector_type(4)))  unsigned int v4u;
typedef __attribute__((ext_vector_type(8)))  unsigned int v8u;

union FragBF16 {
  v16bf v;
  unsigned short u16[16];
  unsigned int   u32[8];
};
union AccF32 {
  v8f   v;
  float f[8];
};

__device__ __forceinline__ unsigned short f32_to_bf16(float f) {
  unsigned int u = __float_as_uint(f);
  u += 0x7FFFu + ((u >> 16) & 1u);  // round-to-nearest-even
  return (unsigned short)(u >> 16);
}

__device__ __forceinline__ unsigned int pack_bf16x2(float lo, float hi) {
  return (unsigned int)f32_to_bf16(lo) | ((unsigned int)f32_to_bf16(hi) << 16);
}

// ---------------------------------------------------------------------------
// Router: logits -> softmax -> top-2 -> renormalized combine weights [T, E]
// ---------------------------------------------------------------------------
__global__ __launch_bounds__(128)
void moe_router(const float* __restrict__ x,
                const float* __restrict__ rw,
                float* __restrict__ combine) {
  const int t = blockIdx.x;
  const int tid = threadIdx.x;
  __shared__ float red[128 * NUM_EXPERTS];
  __shared__ float logits[NUM_EXPERTS];

  float acc[NUM_EXPERTS];
#pragma unroll
  for (int e = 0; e < NUM_EXPERTS; ++e) acc[e] = 0.f;

  const float* xr = x + (size_t)t * HIDDEN;
  for (int k = tid; k < HIDDEN; k += 128) {
    float xv = xr[k];
#pragma unroll
    for (int e = 0; e < NUM_EXPERTS; ++e)
      acc[e] = fmaf(xv, rw[(size_t)e * HIDDEN + k], acc[e]);
  }
#pragma unroll
  for (int e = 0; e < NUM_EXPERTS; ++e) red[tid * NUM_EXPERTS + e] = acc[e];
  __syncthreads();

  if (tid < NUM_EXPERTS) {
    float s = 0.f;
    for (int i = 0; i < 128; ++i) s += red[i * NUM_EXPERTS + tid];
    logits[tid] = s;
  }
  __syncthreads();

  if (tid == 0) {
    float m = logits[0];
#pragma unroll
    for (int e = 1; e < NUM_EXPERTS; ++e) m = fmaxf(m, logits[e]);
    float p[NUM_EXPERTS];
    float sum = 0.f;
#pragma unroll
    for (int e = 0; e < NUM_EXPERTS; ++e) { p[e] = __expf(logits[e] - m); sum += p[e]; }
    float inv = 1.f / sum;
#pragma unroll
    for (int e = 0; e < NUM_EXPERTS; ++e) p[e] *= inv;
    int i1 = 0;
#pragma unroll
    for (int e = 1; e < NUM_EXPERTS; ++e) if (p[e] > p[i1]) i1 = e;
    int i2 = (i1 == 0) ? 1 : 0;
#pragma unroll
    for (int e = 0; e < NUM_EXPERTS; ++e) if (e != i1 && p[e] > p[i2]) i2 = e;
    float norm = 1.f / (p[i1] + p[i2]);
    float* c = combine + (size_t)t * NUM_EXPERTS;
#pragma unroll
    for (int e = 0; e < NUM_EXPERTS; ++e) c[e] = 0.f;
    c[i1] = p[i1] * norm;
    c[i2] = p[i2] * norm;
  }
}

// ---------------------------------------------------------------------------
// Gate+Up GEMM with fused SwiGLU: h = silu(x@Wg) * (x@Wu), bf16 out.
// 128 thr = 4 waves; tile 128(M)x64(N); each wave owns 2 M sub-tiles
// (rows 16*wave and 16*(wave+4)) so every B fragment feeds 2 WMMAs.
// ---------------------------------------------------------------------------
__global__ __launch_bounds__(128)
void moe_gateup(const float* __restrict__ x,
                const float* __restrict__ wg,
                const float* __restrict__ wu,
                unsigned short* __restrict__ h) {
  __shared__ unsigned short As[BM][LDA];
  __shared__ unsigned short Bg[BN][LDA];
  __shared__ unsigned short Bu[BN][LDA];

  const int tid  = threadIdx.x;
  const int lane = tid & 31;
  const int wave = tid >> 5;
  const int m0 = blockIdx.y * BM;
  const int n0 = blockIdx.x * BN;

  AccF32 cg[2][4], cu[2][4];
#pragma unroll
  for (int mt = 0; mt < 2; ++mt)
#pragma unroll
    for (int i = 0; i < 4; ++i)
#pragma unroll
      for (int j = 0; j < 8; ++j) { cg[mt][i].f[j] = 0.f; cu[mt][i].f[j] = 0.f; }

  for (int k0 = 0; k0 < HIDDEN; k0 += KC) {
    // A tile: 128x32 fp32 -> packed bf16 pairs, row-major [m][k]
#pragma unroll
    for (int i = 0; i < 16; ++i) {
      int idx = tid + 128 * i;          // 0..2047 u32 slots
      int m = idx >> 4, k2 = idx & 15;  // 16 u32 per 32-elem row
      const float* src = &x[(size_t)(m0 + m) * HIDDEN + (k0 + 2 * k2)];
      *(unsigned int*)&As[m][2 * k2] = pack_bf16x2(src[0], src[1]);
    }
    // B tiles: 32x64 fp32 -> bf16, stored transposed [n][k]
#pragma unroll
    for (int i = 0; i < 16; ++i) {
      int idx = tid + 128 * i;
      int k = idx >> 6, n = idx & 63;
      size_t g = (size_t)(k0 + k) * INTER + (n0 + n);
      Bg[n][k] = f32_to_bf16(wg[g]);
      Bu[n][k] = f32_to_bf16(wu[g]);
    }
    if (k0 + KC < HIDDEN) {
      size_t pg = (size_t)(k0 + KC + (tid & 31)) * INTER + n0 + ((tid >> 5) << 4);
      __builtin_prefetch(&wg[pg], 0, 1);
      __builtin_prefetch(&wu[pg], 0, 1);
    }
    __syncthreads();

    // Two A fragments per wave (16-bit A 16x32 ISA layout)
    FragBF16 a[2];
#pragma unroll
    for (int mt = 0; mt < 2; ++mt) {
      int row = (wave + 4 * mt) * 16 + (lane & 15);
      int khi = (lane >> 4) << 3;  // 0 or 8
#pragma unroll
      for (int v = 0; v < 8; ++v) {
        int kk = (v < 4) ? (khi + 2 * v) : (16 + khi + 2 * (v - 4));
        a[mt].u32[v] = *(const unsigned int*)&As[row][kk];
      }
    }
#pragma unroll
    for (int nt = 0; nt < 4; ++nt) {
      FragBF16 bg, bu;
      int col = nt * 16 + (lane & 15);
      int klo = (lane >> 4) << 3;  // 0 or 8
#pragma unroll
      for (int v = 0; v < 8; ++v) {
        int kk = (v < 4) ? (klo + 2 * v) : (16 + klo + 2 * (v - 4));
        bg.u32[v] = *(const unsigned int*)&Bg[col][kk];
        bu.u32[v] = *(const unsigned int*)&Bu[col][kk];
      }
#pragma unroll
      for (int mt = 0; mt < 2; ++mt) {
        cg[mt][nt].v = __builtin_amdgcn_wmma_f32_16x16x32_bf16(
            false, a[mt].v, false, bg.v, (short)0, cg[mt][nt].v, false, false);
        cu[mt][nt].v = __builtin_amdgcn_wmma_f32_16x16x32_bf16(
            false, a[mt].v, false, bu.v, (short)0, cu[mt][nt].v, false, false);
      }
    }
    __syncthreads();
  }

  // Epilogue: SwiGLU, store bf16 h. C layout: row = r + (lane>=16 ? 8 : 0)
  const int ccol = lane & 15;
#pragma unroll
  for (int mt = 0; mt < 2; ++mt) {
    const int rbase = m0 + (wave + 4 * mt) * 16 + ((lane >> 4) << 3);
#pragma unroll
    for (int nt = 0; nt < 4; ++nt) {
      int col = n0 + nt * 16 + ccol;
#pragma unroll
      for (int v = 0; v < 8; ++v) {
        float g = cg[mt][nt].f[v];
        float u = cu[mt][nt].f[v];
        float s = g * (1.f / (1.f + __expf(-g)));  // silu
        h[(size_t)(rbase + v) * INTER + col] = f32_to_bf16(s * u);
      }
    }
  }
}

// ---------------------------------------------------------------------------
// Down projection + weighted combine: out (+)= combine[:,e] * (h @ Wd)
// A tile (bf16 h) is DMA'd into LDS by the Tensor Data Mover; its pad
// config (pad every 16 DWORDs by 1 DWORD) reproduces the LDA=34 layout.
// ---------------------------------------------------------------------------
__global__ __launch_bounds__(128)
void moe_down(const unsigned short* __restrict__ h,
              const float* __restrict__ wd,
              const float* __restrict__ combine,
              float* __restrict__ out,
              const int expert) {
  __shared__ unsigned short As[BM][LDA];
  __shared__ unsigned short Bs[BN][LDA];

  const int tid  = threadIdx.x;
  const int lane = tid & 31;
  const int wave = tid >> 5;
  const int m0 = blockIdx.y * BM;
  const int n0 = blockIdx.x * BN;

  AccF32 c[2][4];
#pragma unroll
  for (int mt = 0; mt < 2; ++mt)
#pragma unroll
    for (int i = 0; i < 4; ++i)
#pragma unroll
      for (int j = 0; j < 8; ++j) c[mt][i].f[j] = 0.f;

  const unsigned int lds_as = (unsigned int)(size_t)&As[0][0];

  for (int k0 = 0; k0 < INTER; k0 += KC) {
    // --- A tile via TDM: 2D tile KC x BM of bf16, one op issued by wave 0 ---
    if (wave == 0) {
      unsigned long long ga =
          (unsigned long long)(const void*)h +
          ((unsigned long long)m0 * INTER + (unsigned long long)k0) * 2ull;
      v4u g0;
      g0.x = 1u;                                   // count=1 (valid D#)
      g0.y = lds_as;                               // LDS byte address
      g0.z = (unsigned int)ga;                     // global_addr[31:0]
      g0.w = (unsigned int)(ga >> 32) | (2u << 30);  // global_addr[56:32] | type=2
      v8u g1;
      g1.s0 = (1u << 16)        // data_size = 2 bytes
            | (1u << 20)        // pad_enable
            | (3u << 22);       // pad_interval: every 16 DWORDs (+1 DWORD pad)
      g1.s1 = ((unsigned int)INTER & 0xFFFFu) << 16;   // tensor_dim0[15:0]
      g1.s2 = ((unsigned int)TOKENS & 0xFFFFu) << 16;  // tensor_dim1[15:0]
      g1.s3 = ((unsigned int)KC) << 16;                // tile_dim0 = 32
      g1.s4 = (unsigned int)BM;                        // tile_dim1 = 128
      g1.s5 = (unsigned int)INTER;                     // tensor_dim0_stride lo
      g1.s6 = 0u;
      g1.s7 = 0u;
      v4u g2 = (v4u)0u;  // 2D tensor: groups 2/3 unused (zero-filled)
      v4u g3 = (v4u)0u;
      asm volatile("tensor_load_to_lds %0, %1, %2, %3"
                   : : "s"(g0), "s"(g1), "s"(g2), "s"(g3) : "memory");
    }

    // --- B tile: 32x64 fp32 -> bf16 transposed [n][k] (all waves) ---
#pragma unroll
    for (int i = 0; i < 16; ++i) {
      int idx = tid + 128 * i;
      int k = idx >> 6, n = idx & 63;
      Bs[n][k] = f32_to_bf16(wd[(size_t)(k0 + k) * HIDDEN + (n0 + n)]);
    }
    if (k0 + KC < INTER) {
      size_t pg = (size_t)(k0 + KC + (tid & 31)) * HIDDEN + n0 + ((tid >> 5) << 4);
      __builtin_prefetch(&wd[pg], 0, 1);
    }
    __builtin_amdgcn_s_wait_tensorcnt(0);  // wave 0: TDM done; others: no-op
    __syncthreads();

    FragBF16 a[2];
#pragma unroll
    for (int mt = 0; mt < 2; ++mt) {
      int row = (wave + 4 * mt) * 16 + (lane & 15);
      int khi = (lane >> 4) << 3;
#pragma unroll
      for (int v = 0; v < 8; ++v) {
        int kk = (v < 4) ? (khi + 2 * v) : (16 + khi + 2 * (v - 4));
        a[mt].u32[v] = *(const unsigned int*)&As[row][kk];
      }
    }
#pragma unroll
    for (int nt = 0; nt < 4; ++nt) {
      FragBF16 b;
      int col = nt * 16 + (lane & 15);
      int klo = (lane >> 4) << 3;
#pragma unroll
      for (int v = 0; v < 8; ++v)
        b.u32[v] = *(const unsigned int*)&Bs[col][(v < 4) ? (klo + 2 * v) : (16 + klo + 2 * (v - 4))];
#pragma unroll
      for (int mt = 0; mt < 2; ++mt)
        c[mt][nt].v = __builtin_amdgcn_wmma_f32_16x16x32_bf16(
            false, a[mt].v, false, b.v, (short)0, c[mt][nt].v, false, false);
    }
    __syncthreads();
  }

  const int ccol = lane & 15;
#pragma unroll
  for (int mt = 0; mt < 2; ++mt) {
    const int rbase = m0 + (wave + 4 * mt) * 16 + ((lane >> 4) << 3);
#pragma unroll
    for (int nt = 0; nt < 4; ++nt) {
      int col = n0 + nt * 16 + ccol;
#pragma unroll
      for (int v = 0; v < 8; ++v) {
        int row = rbase + v;
        float s = combine[(size_t)row * NUM_EXPERTS + expert];
        size_t o = (size_t)row * HIDDEN + col;
        float val = s * c[mt][nt].f[v];
        if (expert == 0) out[o] = val;   // overwrite poisoned buffer
        else             out[o] += val;  // accumulate (sequential launches)
      }
    }
  }
}

// ---------------------------------------------------------------------------
extern "C" void kernel_launch(void* const* d_in, const int* in_sizes, int n_in,
                              void* d_out, int out_size, void* d_ws, size_t ws_size,
                              hipStream_t stream) {
  (void)in_sizes; (void)n_in; (void)out_size; (void)ws_size;
  const float* x  = (const float*)d_in[0];
  const float* rw = (const float*)d_in[1];
  const float* wg = (const float*)d_in[2];
  const float* wu = (const float*)d_in[3];
  const float* wd = (const float*)d_in[4];
  float* out = (float*)d_out;

  // ws layout: combine [T,E] f32 (128 KB) | h [T,INTER] bf16 (11.5 MB)
  float* combine = (float*)d_ws;
  unsigned short* h =
      (unsigned short*)((char*)d_ws + (size_t)TOKENS * NUM_EXPERTS * sizeof(float));

  moe_router<<<TOKENS, 128, 0, stream>>>(x, rw, combine);

  for (int e = 0; e < NUM_EXPERTS; ++e) {
    const size_t wsz = (size_t)HIDDEN * INTER;
    moe_gateup<<<dim3(INTER / BN, TOKENS / BM), 128, 0, stream>>>(
        x, wg + (size_t)e * wsz, wu + (size_t)e * wsz, h);
    moe_down<<<dim3(HIDDEN / BN, TOKENS / BM), 128, 0, stream>>>(
        h, wd + (size_t)e * wsz, combine, out, e);
  }
}
